// QLSTMGen113_65481071405283
// MI455X (gfx1250) — matrix operand order repaired and checked by reading, not compiled, not run on hardware
//
#include <hip/hip_runtime.h>

// ---------------------------------------------------------------------------
// LSTM, T=2048 steps, B=64, D_IN=D_H=512.  Persistent-grid kernel:
//   - init kernel packs fp32 weights -> bf16 WMMA-B-fragment layout in d_ws
//   - main kernel: 16 blocks x 256 threads (128 wave32); each wave owns one
//     16x16 tile of the [64,512] gate output and all 4 gates for it.
//   - weights staged once into 256KB LDS/block via global_load_async_to_lds
//   - split-barrier pipeline: x-half GEMM of step t+1 overlapped with the
//     grid barrier wait; only the h-half GEMM + gates are on the serial path.
//   - both GEMM loops software-pipelined by one iteration so global loads
//     overlap WMMA + LDS traffic instead of serializing on s_wait_loadcnt 0.
// ---------------------------------------------------------------------------

#define T_STEPS 2048
#define BATCH   64
#define DIN     512
#define DH      512
#define NBLOCKS 16
#define NTHREADS 256
#define PACKED_DWORDS (4u*32u*32u*32u*8u)   // 1,048,576 dwords = 4MB bf16 weights

typedef __attribute__((ext_vector_type(16))) __bf16 v16bf;
typedef __attribute__((ext_vector_type(8)))  float  v8f;

// fp32 pair -> packed bf16 dword (v_cvt_pk_bf16_f32): lo->[15:0], hi->[31:16]
#if __has_builtin(__builtin_amdgcn_cvt_pk_bf16_f32)
__device__ __forceinline__ unsigned f2bf_pk(float lo, float hi) {
  auto p = __builtin_amdgcn_cvt_pk_bf16_f32(lo, hi);
  return __builtin_bit_cast(unsigned, p);
}
#else
__device__ __forceinline__ unsigned f2bf_pk(float lo, float hi) {
  unsigned r;
  asm("v_cvt_pk_bf16_f32 %0, %1, %2" : "=v"(r) : "v"(lo), "v"(hi));
  return r;
}
#endif
__device__ __forceinline__ unsigned short f2bf(float f) {
  return (unsigned short)f2bf_pk(f, 0.0f);
}

#if __has_builtin(__builtin_amdgcn_tanhf)
__device__ __forceinline__ float fast_tanh(float x) { return __builtin_amdgcn_tanhf(x); }
#else
__device__ __forceinline__ float fast_tanh(float x) { return tanhf(x); }
#endif
__device__ __forceinline__ float fast_sigm(float x) { return 0.5f * (1.0f + fast_tanh(0.5f * x)); }

// ---------------------------------------------------------------------------
// Init: pack W{f,i,u,o} (fp32 [1024,512] row-major) into bf16 B-fragment
// layout: half index = (((nt*4+g)*32+kk)*32+lane)*16 + 2j, where
//   n = nt*16 + (lane&15),  k = kk*32 + (lane>=16 ? 8:0) + 2(j&3) + (j>=4 ? 16:0)  (+0/+1 pair)
// Also zero the initial h buffer (bf16) and the grid-barrier counter.
// ---------------------------------------------------------------------------
__global__ void qlstm_init(const float* __restrict__ Wf, const float* __restrict__ Wi,
                           const float* __restrict__ Wu, const float* __restrict__ Wo,
                           unsigned* __restrict__ packed, unsigned* __restrict__ h0,
                           unsigned* __restrict__ cnt) {
  unsigned idx = blockIdx.x * blockDim.x + threadIdx.x;
  if (idx < PACKED_DWORDS) {
    unsigned j    = idx & 7u;
    unsigned lane = (idx >> 3) & 31u;
    unsigned kk   = (idx >> 8) & 31u;
    unsigned g    = (idx >> 13) & 3u;
    unsigned nt   = idx >> 15;
    unsigned n = nt * 16u + (lane & 15u);
    unsigned k = kk * 32u + ((lane >> 4) << 3) + 2u * (j & 3u) + ((j >> 2) << 4);
    const float* W = (g == 0) ? Wf : (g == 1) ? Wi : (g == 2) ? Wu : Wo;
    float lo = W[(size_t)k * DH + n];
    float hi = W[(size_t)(k + 1) * DH + n];
    packed[idx] = f2bf_pk(lo, hi);
  } else if (idx < PACKED_DWORDS + 16384u) {       // h0: 64*512 bf16 = 16384 dwords
    h0[idx - PACKED_DWORDS] = 0u;
  } else if (idx == PACKED_DWORDS + 16384u) {
    *cnt = 0u;
  }
}

// ---------------------------------------------------------------------------
// Main persistent kernel.
// ---------------------------------------------------------------------------
__global__ void __launch_bounds__(NTHREADS, 1)
qlstm_main(const float* __restrict__ x,
           const float* __restrict__ biasF, const float* __restrict__ biasI,
           const float* __restrict__ biasU, const float* __restrict__ biasO,
           const unsigned* __restrict__ packed,
           unsigned short* __restrict__ h0,   // bf16 ping
           unsigned short* __restrict__ h1,   // bf16 pong
           unsigned* __restrict__ cnt,
           float* __restrict__ out) {
  extern __shared__ char smem[];               // 256KB dynamic LDS: this block's weights

  const unsigned tid     = threadIdx.x;
  const unsigned lane    = tid & 31u;
  const unsigned wave    = tid >> 5;           // 0..7
  const unsigned mt      = wave >> 1;          // 0..3   (batch tile)
  const unsigned ntLocal = wave & 1u;          // 0..1
  const unsigned nt      = blockIdx.x * 2u + ntLocal;
  const unsigned m0      = mt * 16u;
  const unsigned n0      = nt * 16u;
  const unsigned nIdx    = lane & 15u;         // N (and A-row M) within tile
  const unsigned hiHalf  = lane >> 4;
  const unsigned moff    = hiHalf * 8u;        // C/D row offset for lanes 16-31
  const unsigned kbase   = hiHalf * 8u;        // A-fragment K base
  const unsigned aRow    = m0 + nIdx;          // global batch row for A loads
  const unsigned ldsWaveBase = ntLocal * 131072u;  // [ntLocal][g][kk][lane][16 half]

  // ---- Stage this block's 256KB of packed bf16 weights into LDS (async path)
  {
    unsigned long long gsrc =
        (unsigned long long)((const char*)packed + (size_t)blockIdx.x * 262144u);
    unsigned ldsBase = (unsigned)(size_t)(void*)smem;   // low 32 bits = LDS byte addr
    for (unsigned i = 0; i < 64u; ++i) {
      unsigned off = (i * NTHREADS + tid) * 16u;        // 16B per lane per issue
      unsigned ldsAddr = ldsBase + off;
      asm volatile("global_load_async_to_lds_b128 %0, %1, %2"
                   :: "v"(ldsAddr), "v"(off), "s"(gsrc) : "memory");
    }
    asm volatile("s_wait_asynccnt 0" ::: "memory");
    __syncthreads();
  }

  // Per-wave LDS base for B fragments: tile (g,kk) at bwave + g*32768 + kk*1024
  const char* bwave = smem + ldsWaveBase + lane * 32u;

  // ---- Per-lane persistent state
  float c[8];
#pragma unroll
  for (int r = 0; r < 8; ++r) c[r] = 0.0f;

  const float bvF = biasF[n0 + nIdx], bvI = biasI[n0 + nIdx];
  const float bvU = biasU[n0 + nIdx], bvO = biasO[n0 + nIdx];

  v8f accF, accI, accU, accO;

  // X_HALF(tt): acc* = bias + x_tt-contribution (K = 0..511).  Runs in the
  // barrier shadow.  Software-pipelined: next iteration's 4 global loads are
  // issued before the current iteration's converts/WMMAs consume theirs.
#define X_HALF(tt)                                                               \
  {                                                                              \
    accF = (v8f){bvF, bvF, bvF, bvF, bvF, bvF, bvF, bvF};                        \
    accI = (v8f){bvI, bvI, bvI, bvI, bvI, bvI, bvI, bvI};                        \
    accU = (v8f){bvU, bvU, bvU, bvU, bvU, bvU, bvU, bvU};                        \
    accO = (v8f){bvO, bvO, bvO, bvO, bvO, bvO, bvO, bvO};                        \
    const float* xr = x + ((size_t)(tt) * BATCH + aRow) * DIN + kbase;           \
    __builtin_prefetch(xr + BATCH * DIN, 0, 1);                                  \
    float4 c0 = *(const float4*)(xr);                                            \
    float4 c1 = *(const float4*)(xr + 4);                                        \
    float4 c2 = *(const float4*)(xr + 16);                                       \
    float4 c3 = *(const float4*)(xr + 20);                                       \
    _Pragma("unroll 4")                                                          \
    for (unsigned kk = 0; kk < 16u; ++kk) {                                      \
      const float* pn = xr + ((kk + 1u) & 15u) * 32u;                            \
      float4 n0_ = *(const float4*)(pn);                                         \
      float4 n1_ = *(const float4*)(pn + 4);                                     \
      float4 n2_ = *(const float4*)(pn + 16);                                    \
      float4 n3_ = *(const float4*)(pn + 20);                                    \
      union { v16bf v; unsigned u[8]; } A;                                       \
      A.u[0] = f2bf_pk(c0.x, c0.y); A.u[1] = f2bf_pk(c0.z, c0.w);                \
      A.u[2] = f2bf_pk(c1.x, c1.y); A.u[3] = f2bf_pk(c1.z, c1.w);                \
      A.u[4] = f2bf_pk(c2.x, c2.y); A.u[5] = f2bf_pk(c2.z, c2.w);                \
      A.u[6] = f2bf_pk(c3.x, c3.y); A.u[7] = f2bf_pk(c3.z, c3.w);                \
      const char* bb = bwave + kk * 1024u;                                       \
      v16bf Bf = *(const v16bf*)(bb);                                            \
      v16bf Bi = *(const v16bf*)(bb + 32768u);                                   \
      v16bf Bu = *(const v16bf*)(bb + 65536u);                                   \
      v16bf Bo = *(const v16bf*)(bb + 98304u);                                   \
      accF = __builtin_amdgcn_wmma_f32_16x16x32_bf16(false, A.v, false, Bf, (short)0, accF, false, false); \
      accI = __builtin_amdgcn_wmma_f32_16x16x32_bf16(false, A.v, false, Bi, (short)0, accI, false, false); \
      accU = __builtin_amdgcn_wmma_f32_16x16x32_bf16(false, A.v, false, Bu, (short)0, accU, false, false); \
      accO = __builtin_amdgcn_wmma_f32_16x16x32_bf16(false, A.v, false, Bo, (short)0, accO, false, false); \
      c0 = n0_; c1 = n1_; c2 = n2_; c3 = n3_;                                    \
    }                                                                            \
  }

  X_HALF(0u);   // prologue: x-contribution of step 0

  for (unsigned t = 0; t < T_STEPS; ++t) {
    const unsigned short* hsrc = (t & 1u) ? h1 : h0;
    unsigned short*       hdst = (t & 1u) ? h0 : h1;

    // ---- h-half (serial critical path): K 512..1023, A = h_{t-1} (bf16).
    // Fully unrolled + software-pipelined by one iteration.
    {
      const unsigned* hb = (const unsigned*)(hsrc) + (aRow * DH + kbase) / 2u;
      uint4 q0 = *(const uint4*)(hb);
      uint4 q1 = *(const uint4*)(hb + 8);
#pragma unroll
      for (unsigned kk = 0; kk < 16u; ++kk) {
        const unsigned* hn = hb + ((kk + 1u) & 15u) * 16u;
        uint4 r0 = *(const uint4*)(hn);
        uint4 r1 = *(const uint4*)(hn + 8);
        union { v16bf v; unsigned u[8]; } A;
        A.u[0] = q0.x; A.u[1] = q0.y; A.u[2] = q0.z; A.u[3] = q0.w;
        A.u[4] = q1.x; A.u[5] = q1.y; A.u[6] = q1.z; A.u[7] = q1.w;
        const char* bb = bwave + (kk + 16u) * 1024u;
        v16bf Bf = *(const v16bf*)(bb);
        v16bf Bi = *(const v16bf*)(bb + 32768u);
        v16bf Bu = *(const v16bf*)(bb + 65536u);
        v16bf Bo = *(const v16bf*)(bb + 98304u);
        accF = __builtin_amdgcn_wmma_f32_16x16x32_bf16(false, A.v, false, Bf, (short)0, accF, false, false);
        accI = __builtin_amdgcn_wmma_f32_16x16x32_bf16(false, A.v, false, Bi, (short)0, accI, false, false);
        accU = __builtin_amdgcn_wmma_f32_16x16x32_bf16(false, A.v, false, Bu, (short)0, accU, false, false);
        accO = __builtin_amdgcn_wmma_f32_16x16x32_bf16(false, A.v, false, Bo, (short)0, accO, false, false);
        q0 = r0; q1 = r1;
      }
    }

    // ---- Gate math (hw tanh), update c in registers, emit h
#pragma unroll
    for (int r = 0; r < 8; ++r) {
      float fg = fast_sigm(accF[r]);
      float ig = fast_sigm(accI[r]);
      float gg = fast_tanh(accU[r]);
      float og = fast_sigm(accO[r]);
      float cn = fg * c[r] + ig * gg;
      c[r] = cn;
      float hn = og * fast_tanh(cn);
      unsigned row = m0 + moff + (unsigned)r;
      __builtin_nontemporal_store(hn, &out[((size_t)t * BATCH + row) * DH + n0 + nIdx]);
      hdst[row * DH + n0 + nIdx] = f2bf(hn);                    // recurrent state
      if (t == T_STEPS - 1) {
        size_t hxOff = (size_t)T_STEPS * BATCH * DH;
        out[hxOff + row * DH + n0 + nIdx] = hn;                 // hx
        out[hxOff + (size_t)BATCH * DH + row * DH + n0 + nIdx] = cn;  // cx
      }
    }

    if (t + 1u < T_STEPS) {
      // ---- Split grid barrier: arrive, overlap next x-half, then wait.
      __threadfence();                 // publish h_t (device scope)
      __syncthreads();
      if (tid == 0)
        __hip_atomic_fetch_add(cnt, 1u, __ATOMIC_RELEASE, __HIP_MEMORY_SCOPE_AGENT);

      X_HALF(t + 1u);                  // barrier-shadow work (x only, no h dep)

      if (tid == 0) {
        unsigned target = (t + 1u) * NBLOCKS;
        while (__hip_atomic_load(cnt, __ATOMIC_ACQUIRE, __HIP_MEMORY_SCOPE_AGENT) < target)
          __builtin_amdgcn_s_sleep(1);
      }
      __syncthreads();
      __threadfence();                 // acquire: invalidate so h_t reads are fresh
    }
  }
#undef X_HALF
}

// ---------------------------------------------------------------------------
extern "C" void kernel_launch(void* const* d_in, const int* in_sizes, int n_in,
                              void* d_out, int out_size, void* d_ws, size_t ws_size,
                              hipStream_t stream) {
  (void)in_sizes; (void)n_in; (void)out_size; (void)ws_size;
  const float* x  = (const float*)d_in[0];
  const float* Wf = (const float*)d_in[1];
  const float* bf = (const float*)d_in[2];
  const float* Wi = (const float*)d_in[3];
  const float* bi = (const float*)d_in[4];
  const float* Wu = (const float*)d_in[5];
  const float* bu = (const float*)d_in[6];
  const float* Wo = (const float*)d_in[7];
  const float* bo = (const float*)d_in[8];
  float* out = (float*)d_out;

  char* ws = (char*)d_ws;
  unsigned*       packed = (unsigned*)ws;                            // 4 MB
  unsigned short* h0     = (unsigned short*)(ws + (4u << 20));       // 64 KB
  unsigned short* h1     = (unsigned short*)(ws + (4u << 20) + (64u << 10));
  unsigned*       cnt    = (unsigned*)(ws + (4u << 20) + (128u << 10));

  unsigned initN = PACKED_DWORDS + 16384u + 1u;
  qlstm_init<<<(initN + 255u) / 256u, 256, 0, stream>>>(Wf, Wi, Wu, Wo, packed,
                                                        (unsigned*)h0, cnt);
  qlstm_main<<<NBLOCKS, NTHREADS, 262144, stream>>>(x, bf, bi, bu, bo, packed,
                                                    h0, h1, cnt, out);
}
